// GATv2_36215164240054
// MI455X (gfx1250) — matrix operand order repaired and checked by reading, not compiled, not run on hardware
//
#include <hip/hip_runtime.h>
#include <hip/hip_bf16.h>

#define N_NODES   50000
#define N_EDGES   800000
#define D_FEAT    128
#define EMBED     128
#define NUM_HEADS 8
#define HEAD_DIM  16

typedef __attribute__((ext_vector_type(16))) __bf16 v16bf;
typedef __attribute__((ext_vector_type(8)))  __bf16 v8bf;
typedef __attribute__((ext_vector_type(8)))  float  v8f;

// ---------------------------------------------------------------------------
// float atomic-max via monotonic int punning (init value must be -inf bits)
// ---------------------------------------------------------------------------
__device__ __forceinline__ void atomicMaxF(float* addr, float val) {
  int iv = __float_as_int(val);
  if (iv >= 0) {
    atomicMax((int*)addr, iv);
  } else {
    atomicMin((unsigned int*)addr, (unsigned int)iv);
  }
}

// ---------------------------------------------------------------------------
// Convert nodes + weight matrices to bf16 (RNE)
// ---------------------------------------------------------------------------
__global__ void k_prep(const float* __restrict__ nodes,
                       const float* __restrict__ Ws, const float* __restrict__ Wr,
                       __bf16* __restrict__ nodes_bf,
                       __bf16* __restrict__ Ws_bf, __bf16* __restrict__ Wr_bf) {
  int i = blockIdx.x * blockDim.x + threadIdx.x;
  if (i < N_NODES * D_FEAT) nodes_bf[i] = (__bf16)nodes[i];
  if (i < D_FEAT * EMBED) {
    Ws_bf[i] = (__bf16)Ws[i];
    Wr_bf[i] = (__bf16)Wr[i];
  }
}

__global__ void k_init(float* __restrict__ out, float* __restrict__ nmax,
                       float* __restrict__ denom) {
  int i = blockIdx.x * blockDim.x + threadIdx.x;
  if (i < N_NODES * EMBED) out[i] = 0.0f;
  if (i < N_NODES * NUM_HEADS) {
    nmax[i]  = __uint_as_float(0xFF800000u);  // -inf
    denom[i] = 0.0f;
  }
}

// ---------------------------------------------------------------------------
// Dual GEMM: Ps = A*Bs + bias_s, Pr = A*Br + bias_r  (bf16 in, f32 out)
// One wave owns one 16x16 output tile; 8 waves/block cover N=128;
// grid.x covers M = 50000 (3125 blocks * 16 rows).
// Fragment layouts per CDNA5 ISA 7.12.2 (wave32).
// ---------------------------------------------------------------------------
__global__ __launch_bounds__(256) void k_gemm(
    const __bf16* __restrict__ A,   // [N_NODES][128] bf16
    const __bf16* __restrict__ Bs,  // [128][128] bf16 (K-major)
    const __bf16* __restrict__ Br,  // [128][128] bf16
    const float*  __restrict__ bs,  // [128] bias (Ws_b flattened)
    const float*  __restrict__ br,  // [128] bias (Wr_b flattened)
    float* __restrict__ Ps, float* __restrict__ Pr) {
  const int lane = threadIdx.x & 31;
  const int wave = threadIdx.x >> 5;
  const int m0   = blockIdx.x * 16;
  const int n0   = wave * 16;

  // A fragment: lane&15 = row, lane>>4 selects K sub-blocks {0..7,16..23} vs {8..15,24..31}
  const int mrow = m0 + (lane & 15);
  const int koff = (lane >> 4) * 8;

  v8f accS = {};
  v8f accR = {};

#pragma unroll
  for (int k0 = 0; k0 < D_FEAT; k0 += 32) {
    const __bf16* ap = A + (size_t)mrow * D_FEAT + (k0 + koff);
    v8bf alo = *(const v8bf*)(ap);        // K = k0+koff .. +7
    v8bf ahi = *(const v8bf*)(ap + 16);   // K = k0+koff+16 .. +23
    v16bf afrag = __builtin_shufflevector(alo, ahi,
        0, 1, 2, 3, 4, 5, 6, 7, 8, 9, 10, 11, 12, 13, 14, 15);

    // B fragment: lane&15 = N column, elements over K in same order as A fragment
    const int ncol = n0 + (lane & 15);
    v16bf bsf, brf;
#pragma unroll
    for (int s = 0; s < 16; ++s) {
      const int kk = k0 + koff + (s < 8 ? s : s + 8);
      bsf[s] = Bs[(size_t)kk * EMBED + ncol];
      brf[s] = Br[(size_t)kk * EMBED + ncol];
    }

    accS = __builtin_amdgcn_wmma_f32_16x16x32_bf16(
        false, afrag, false, bsf, (short)0, accS, false, false);
    accR = __builtin_amdgcn_wmma_f32_16x16x32_bf16(
        false, afrag, false, brf, (short)0, accR, false, false);
  }

  // C/D layout: lanes 0-15 hold M=i, lanes 16-31 hold M=8+i in VGPR i; N = lane&15
  const int row0 = m0 + (lane >> 4) * 8;
  const int col  = n0 + (lane & 15);
  const float biasS = bs[col];
  const float biasR = br[col];
#pragma unroll
  for (int i = 0; i < 8; ++i) {
    Ps[(size_t)(row0 + i) * EMBED + col] = accS[i] + biasS;
    Pr[(size_t)(row0 + i) * EMBED + col] = accR[i] + biasR;
  }
}

// ---------------------------------------------------------------------------
// Pass 1 over edges: z = Ps[s] + Pr[r]; x = mish(z); logit[h] = x . attn_k + b
// wave per edge; lane handles dims 4*lane..4*lane+3 (head = lane>>2)
// ---------------------------------------------------------------------------
__global__ __launch_bounds__(256) void k_edge_logits(
    const int* __restrict__ senders, const int* __restrict__ receivers,
    const float* __restrict__ Ps, const float* __restrict__ Pr,
    const float* __restrict__ attn_k, const float* __restrict__ attn_b,
    float* __restrict__ logits, float* __restrict__ nmax) {
  const int lane = threadIdx.x & 31;
  const int e = blockIdx.x * 8 + (threadIdx.x >> 5);
  if (e >= N_EDGES) return;
  const int s = senders[e];
  const int r = receivers[e];

  const float4 ps = ((const float4*)(Ps + (size_t)s * EMBED))[lane];
  const float4 pr = ((const float4*)(Pr + (size_t)r * EMBED))[lane];

  const int dh = (4 * lane) & 15;  // position within head
  float ak0 = attn_k[dh + 0], ak1 = attn_k[dh + 1];
  float ak2 = attn_k[dh + 2], ak3 = attn_k[dh + 3];

  float z[4] = {ps.x + pr.x, ps.y + pr.y, ps.z + pr.z, ps.w + pr.w};
  float ak[4] = {ak0, ak1, ak2, ak3};
  float p = 0.0f;
#pragma unroll
  for (int j = 0; j < 4; ++j) {
    float x  = z[j];
    float sp = log1pf(expf(x));   // softplus
    float mh = x * tanhf(sp);     // mish
    p += mh * ak[j];
  }
  // reduce over the 4 lanes of this head
  p += __shfl_xor(p, 1, 32);
  p += __shfl_xor(p, 2, 32);

  if ((lane & 3) == 0) {
    const int h = lane >> 2;
    const float lg = p + attn_b[0];
    logits[(size_t)e * NUM_HEADS + h] = lg;
    atomicMaxF(&nmax[(size_t)r * NUM_HEADS + h], lg);
  }
}

// ---------------------------------------------------------------------------
// Pass 2: u = exp(logit - seg_max); denom += u   (thread per edge*head)
// ---------------------------------------------------------------------------
__global__ void k_edge_exp(const int* __restrict__ receivers,
                           const float* __restrict__ nmax,
                           float* __restrict__ logits,
                           float* __restrict__ denom) {
  int t = blockIdx.x * blockDim.x + threadIdx.x;
  if (t >= N_EDGES * NUM_HEADS) return;
  const int e = t >> 3;
  const int h = t & 7;
  const int r = receivers[e];
  const float u = expf(logits[t] - nmax[(size_t)r * NUM_HEADS + h]);
  logits[t] = u;
  atomicAdd(&denom[(size_t)r * NUM_HEADS + h], u);
}

// ---------------------------------------------------------------------------
// Pass 3: out[r] += Ps[s] * (u / denom[r])   (wave per edge, atomics into L2)
// ---------------------------------------------------------------------------
__global__ __launch_bounds__(256) void k_edge_agg(
    const int* __restrict__ senders, const int* __restrict__ receivers,
    const float* __restrict__ Ps, const float* __restrict__ logits,
    const float* __restrict__ denom, float* __restrict__ out) {
  const int lane = threadIdx.x & 31;
  const int e = blockIdx.x * 8 + (threadIdx.x >> 5);
  if (e >= N_EDGES) return;
  const int s = senders[e];
  const int r = receivers[e];
  const int h = lane >> 2;

  const float w = logits[(size_t)e * NUM_HEADS + h] /
                  denom[(size_t)r * NUM_HEADS + h];
  const float4 ps = ((const float4*)(Ps + (size_t)s * EMBED))[lane];
  float* o = out + (size_t)r * EMBED + 4 * lane;
  atomicAdd(o + 0, ps.x * w);
  atomicAdd(o + 1, ps.y * w);
  atomicAdd(o + 2, ps.z * w);
  atomicAdd(o + 3, ps.w * w);
}

// ---------------------------------------------------------------------------
extern "C" void kernel_launch(void* const* d_in, const int* in_sizes, int n_in,
                              void* d_out, int out_size, void* d_ws, size_t ws_size,
                              hipStream_t stream) {
  const float* nodes     = (const float*)d_in[0];
  const int*   senders   = (const int*)d_in[1];
  const int*   receivers = (const int*)d_in[2];
  const float* Ws_k      = (const float*)d_in[3];
  const float* Ws_b      = (const float*)d_in[4];
  const float* Wr_k      = (const float*)d_in[5];
  const float* Wr_b      = (const float*)d_in[6];
  const float* attn_k    = (const float*)d_in[7];
  const float* attn_b    = (const float*)d_in[8];
  float* out = (float*)d_out;

  char* ws = (char*)d_ws;
  size_t off = 0;
  auto take = [&](size_t bytes) -> char* {
    char* p = ws + off;
    off += (bytes + 255) & ~(size_t)255;
    return p;
  };
  __bf16* nodes_bf = (__bf16*)take((size_t)N_NODES * D_FEAT * 2);
  __bf16* Ws_bf    = (__bf16*)take((size_t)D_FEAT * EMBED * 2);
  __bf16* Wr_bf    = (__bf16*)take((size_t)D_FEAT * EMBED * 2);
  float*  Ps       = (float*)take((size_t)N_NODES * EMBED * 4);
  float*  Pr       = (float*)take((size_t)N_NODES * EMBED * 4);
  float*  logits   = (float*)take((size_t)N_EDGES * NUM_HEADS * 4);
  float*  nmax     = (float*)take((size_t)N_NODES * NUM_HEADS * 4);
  float*  denom    = (float*)take((size_t)N_NODES * NUM_HEADS * 4);

  const int nconv = (N_NODES * D_FEAT + 255) / 256;          // 25000
  k_prep<<<nconv, 256, 0, stream>>>(nodes, Ws_k, Wr_k, nodes_bf, Ws_bf, Wr_bf);
  k_init<<<(N_NODES * EMBED + 255) / 256, 256, 0, stream>>>(out, nmax, denom);
  k_gemm<<<N_NODES / 16, 256, 0, stream>>>(nodes_bf, Ws_bf, Wr_bf, Ws_b, Wr_b, Ps, Pr);
  k_edge_logits<<<N_EDGES / 8, 256, 0, stream>>>(senders, receivers, Ps, Pr,
                                                 attn_k, attn_b, logits, nmax);
  k_edge_exp<<<(N_EDGES * NUM_HEADS + 255) / 256, 256, 0, stream>>>(receivers, nmax,
                                                                    logits, denom);
  k_edge_agg<<<N_EDGES / 8, 256, 0, stream>>>(senders, receivers, Ps, logits, denom, out);
}
